// CosmicPiMoE_44942537785604
// MI455X (gfx1250) — matrix operand order repaired and checked by reading, not compiled, not run on hardware
//
#include <hip/hip_runtime.h>

// ---------------------------------------------------------------------------
// CDNA5 (gfx1250) MoE forward: bf16 WMMA GEMMs + flash attention,
// async global->LDS tile streaming (ASYNCcnt) with sync fallback.
// ---------------------------------------------------------------------------

typedef __attribute__((ext_vector_type(16))) __bf16 v16bf;
typedef __attribute__((ext_vector_type(8)))  float  v8f;
typedef __attribute__((ext_vector_type(4)))  int    i32x4;

union FragBF { v16bf v; unsigned int u[8]; };

#define AS1 __attribute__((address_space(1)))
#define AS3 __attribute__((address_space(3)))

#if defined(__AMDGCN__) && __has_builtin(__builtin_amdgcn_global_load_async_to_lds_b128) && __has_builtin(__builtin_amdgcn_s_wait_asynccnt)
#define CDNA5_HAS_ASYNC 1
#else
#define CDNA5_HAS_ASYNC 0
#endif

__device__ __forceinline__ unsigned short f2bf(float f) {
  unsigned int u = __builtin_bit_cast(unsigned int, f);
  u += 0x7FFFu + ((u >> 16) & 1u);          // round-to-nearest-even
  return (unsigned short)(u >> 16);
}
__device__ __forceinline__ unsigned int pack_bf16(float a, float b) {
  return (unsigned int)f2bf(a) | ((unsigned int)f2bf(b) << 16);
}

static constexpr int I_ = 1024, D_ = 2048, E_ = 4, B_ = 4, S_ = 256, H_ = 64;
static constexpr int M_ = B_ * S_;  // 1024 token rows per expert

// ---------------------------------------------------------------------------
// f32 -> raw bf16 (activations)
// ---------------------------------------------------------------------------
__global__ __launch_bounds__(256) void f32_to_bf16_kernel(
    const float* __restrict__ s, unsigned short* __restrict__ d, long long n) {
  long long i = (long long)blockIdx.x * 256 + threadIdx.x;
  if (i < n) d[i] = f2bf(s[i]);
}

// ---------------------------------------------------------------------------
// f32 [E,K,N] -> bf16 k-pair-packed uint [E,K/2,N]:
// d[(e*K/2+kp)*N + n] = {bf16(src[2kp][n]), bf16(src[2kp+1][n])}
// (e,kp) fold into one row index since (e*K/2+kp)*2N + n == e*K*N + 2kp*N + n.
// ---------------------------------------------------------------------------
__global__ __launch_bounds__(256) void f32_to_bf16pk_kernel(
    const float* __restrict__ s, unsigned int* __restrict__ d,
    int N, long long nPairs) {
  long long j = (long long)blockIdx.x * 256 + threadIdx.x;
  if (j >= nPairs) return;
  long long rowp = j / N;
  int n = (int)(j - rowp * N);
  const float* p = s + rowp * 2 * N + n;
  d[j] = pack_bf16(p[0], p[N]);
}

// ---------------------------------------------------------------------------
// GEMM: out[e] = act(A[e] @ W[e] + bias[e]); A bf16 row-major, W k-pair-packed.
// EPI: 0 = bias -> f32 ; 1 = relu -> bf16 ; 2 = bias -> bf16 ;
//      3 = sigmoid -> f32 ; 4 = tanh -> f32
// Tile 128x128x32, 8 waves (wave32), wave grid 4(M) x 2(N),
// double-buffered LDS fed by GLOBAL_LOAD_ASYNC_TO_LDS_B128 (ASYNCcnt).
// ---------------------------------------------------------------------------
template <int EPI>
__global__ __launch_bounds__(256) void gemm_bf16_wmma(
    const unsigned short* __restrict__ A, long long aStrideE,
    const unsigned int* __restrict__ W,   // [E][K/2][N] packed bf16x2
    const float* __restrict__ bias,
    float* __restrict__ outF, unsigned short* __restrict__ outB,
    int M, int N, int K) {
  const int e  = blockIdx.z;
  const int bm = blockIdx.y * 128;
  const int bn = blockIdx.x * 128;
  const int tid  = threadIdx.x;
  const int lane = tid & 31;
  const int wv   = tid >> 5;
  const int l15  = lane & 15;
  const int hi   = lane >> 4;
  const int wm   = wv >> 1;   // 0..3
  const int wn   = wv & 1;    // 0..1

  const unsigned short* Ae = A + (size_t)e * (size_t)aStrideE;
  const unsigned int*   We = W + (size_t)e * ((size_t)K / 2) * N;
  const float*          be = bias + (size_t)e * N;

  __shared__ __align__(16) unsigned short As[2][128][40];  // 80B rows: aligned+conflict-free
  __shared__ __align__(16) unsigned int   Bpk[2][16][128];

  v8f acc[2][4];
#pragma unroll
  for (int mi = 0; mi < 2; ++mi)
#pragma unroll
    for (int ni = 0; ni < 4; ++ni)
#pragma unroll
      for (int r = 0; r < 8; ++r) acc[mi][ni][r] = 0.0f;

  const int arow = tid >> 1, ahalf = tid & 1;    // A tile: 128 rows x 32 bf16
  const int kp = tid >> 4, cs = (tid & 15) * 8;  // B tile: 16 kp x 128 uints
  const int nK = K / 32;

  auto compute = [&](int buf) {
    FragBF af[2];
#pragma unroll
    for (int mi = 0; mi < 2; ++mi) {
      const int m = wm * 32 + mi * 16 + l15;
#pragma unroll
      for (int v = 0; v < 8; ++v) {
        const int k = 2 * v + 16 * hi;
        af[mi].u[v] = *(const unsigned int*)&As[buf][m][k];
      }
    }
#pragma unroll
    for (int ni = 0; ni < 4; ++ni) {
      FragBF bf;
      const int n = wn * 64 + ni * 16 + l15;
#pragma unroll
      for (int v = 0; v < 8; ++v) bf.u[v] = Bpk[buf][v + 8 * hi][n];
#pragma unroll
      for (int mi = 0; mi < 2; ++mi)
        acc[mi][ni] = __builtin_amdgcn_wmma_f32_16x16x32_bf16(
            false, af[mi].v, false, bf.v, (short)0, acc[mi][ni], false, false);
    }
  };

#if CDNA5_HAS_ASYNC
  auto issue = [&](int k0, int buf) {
    const unsigned short* aSrc = Ae + (size_t)(bm + arow) * K + k0 + ahalf * 16;
    const unsigned int*   bSrc = We + (size_t)(k0 / 2 + kp) * N + bn + cs;
    unsigned short* aDst = &As[buf][arow][ahalf * 16];
    unsigned int*   bDst = &Bpk[buf][kp][cs];
    __builtin_amdgcn_global_load_async_to_lds_b128(
        (AS1 i32x4*)(aSrc), (AS3 i32x4*)(aDst), 0, 0);
    __builtin_amdgcn_global_load_async_to_lds_b128(
        (AS1 i32x4*)(aSrc + 8), (AS3 i32x4*)(aDst + 8), 0, 0);
    __builtin_amdgcn_global_load_async_to_lds_b128(
        (AS1 i32x4*)(bSrc), (AS3 i32x4*)(bDst), 0, 0);
    __builtin_amdgcn_global_load_async_to_lds_b128(
        (AS1 i32x4*)(bSrc + 4), (AS3 i32x4*)(bDst + 4), 0, 0);
  };
  issue(0, 0);
  for (int kt = 0; kt < nK; ++kt) {
    const int buf = kt & 1;
    if (kt + 1 < nK) {
      issue((kt + 1) * 32, buf ^ 1);            // stream next tile
      __builtin_amdgcn_s_wait_asynccnt(4);      // current tile resident
    } else {
      __builtin_amdgcn_s_wait_asynccnt(0);
    }
    __syncthreads();
    compute(buf);
    __syncthreads();                            // buffer reuse fence
  }
#else
  for (int kt = 0; kt < nK; ++kt) {
    const int k0 = kt * 32;
    const unsigned short* aSrc = Ae + (size_t)(bm + arow) * K + k0 + ahalf * 16;
    uint4 a0 = ((const uint4*)aSrc)[0];
    uint4 a1 = ((const uint4*)aSrc)[1];
    const unsigned int* bSrc = We + (size_t)(k0 / 2 + kp) * N + bn + cs;
    uint4 b0 = ((const uint4*)bSrc)[0];
    uint4 b1 = ((const uint4*)bSrc)[1];
    if (k0 + 32 < K) {
      __builtin_prefetch(aSrc + 32, 0, 0);
      __builtin_prefetch(bSrc + (size_t)16 * N, 0, 0);
    }
    uint4* ad = (uint4*)&As[0][arow][ahalf * 16];
    ad[0] = a0; ad[1] = a1;
    uint4* bd = (uint4*)&Bpk[0][kp][cs];
    bd[0] = b0; bd[1] = b1;
    __syncthreads();
    compute(0);
    __syncthreads();
  }
#endif

  // ---- epilogue: bias + activation + store ----
#pragma unroll
  for (int mi = 0; mi < 2; ++mi)
#pragma unroll
    for (int ni = 0; ni < 4; ++ni)
#pragma unroll
      for (int r = 0; r < 8; ++r) {
        const int row = bm + wm * 32 + mi * 16 + r + 8 * hi;
        const int col = bn + wn * 64 + ni * 16 + l15;
        const size_t idx = ((size_t)e * M + row) * N + col;
        float v = acc[mi][ni][r] + be[col];
        if constexpr (EPI == 1) v = fmaxf(v, 0.0f);
        if constexpr (EPI == 3) v = 1.0f / (1.0f + __expf(-v));
        if constexpr (EPI == 4) v = tanhf(v);
        if constexpr (EPI == 0 || EPI == 3 || EPI == 4) outF[idx] = v;
        else outB[idx] = f2bf(v);
      }
}

// ---------------------------------------------------------------------------
// LayerNorm over last dim (N), per-expert gamma/beta.
// ---------------------------------------------------------------------------
template <bool OUTBF>
__global__ __launch_bounds__(256) void layernorm_kernel(
    const float* __restrict__ in, const float* __restrict__ g,
    const float* __restrict__ b, float* __restrict__ outF,
    unsigned short* __restrict__ outB, int N, int rowsPerExpert) {
  const int row = blockIdx.x;
  const int e = row / rowsPerExpert;
  const float* r  = in + (size_t)row * N;
  const float* ge = g + (size_t)e * N;
  const float* be = b + (size_t)e * N;
  __shared__ float s1[256], s2[256];
  float a = 0.f, q = 0.f;
  for (int i = threadIdx.x; i < N; i += 256) { float v = r[i]; a += v; q += v * v; }
  s1[threadIdx.x] = a; s2[threadIdx.x] = q;
  __syncthreads();
  for (int st = 128; st > 0; st >>= 1) {
    if ((int)threadIdx.x < st) {
      s1[threadIdx.x] += s1[threadIdx.x + st];
      s2[threadIdx.x] += s2[threadIdx.x + st];
    }
    __syncthreads();
  }
  const float mu  = s1[0] / N;
  const float var = s2[0] / N - mu * mu;
  const float rstd = rsqrtf(var + 1e-5f);
  for (int i = threadIdx.x; i < N; i += 256) {
    float v = (r[i] - mu) * rstd * ge[i] + be[i];
    if (OUTBF) outB[(size_t)row * N + i] = f2bf(v);
    else       outF[(size_t)row * N + i] = v;
  }
}

// ---------------------------------------------------------------------------
// Flash attention per (e,b,h): S=256 keys, DH=32, bf16 qkv, f32 softmax.
// K/V rows streamed to LDS with async-b128; 8 waves; wave w owns 32 queries.
// ---------------------------------------------------------------------------
__global__ __launch_bounds__(256) void attention_kernel(
    const unsigned short* __restrict__ qkv, unsigned short* __restrict__ ctxOut) {
  const int h = blockIdx.x, b = blockIdx.y, e = blockIdx.z;
  const int tid = threadIdx.x;
  const int lane = tid & 31, w = tid >> 5;
  const int l15 = lane & 15, hi = lane >> 4;
  const size_t row0 = (size_t)e * M_ + (size_t)b * S_;
  const unsigned short* base = qkv + row0 * (size_t)(3 * D_);
  const int qoff = h * 32, koff = D_ + h * 32, voff = 2 * D_ + h * 32;

  __shared__ __align__(16) unsigned short ks[256][40];  // K rows (bf16, padded)
  __shared__ __align__(16) unsigned short vs[256][40];  // V rows (bf16, padded)
  __shared__ unsigned short Pbuf[8][32][34];

#if CDNA5_HAS_ASYNC
  {
    const unsigned short* ksrc = base + (size_t)tid * (3 * D_) + koff;
    const unsigned short* vsrc = base + (size_t)tid * (3 * D_) + voff;
    unsigned short* kdst = &ks[tid][0];
    unsigned short* vdst = &vs[tid][0];
#pragma unroll
    for (int c = 0; c < 4; ++c) {
      __builtin_amdgcn_global_load_async_to_lds_b128(
          (AS1 i32x4*)(ksrc + 8 * c), (AS3 i32x4*)(kdst + 8 * c), 0, 0);
      __builtin_amdgcn_global_load_async_to_lds_b128(
          (AS1 i32x4*)(vsrc + 8 * c), (AS3 i32x4*)(vdst + 8 * c), 0, 0);
    }
  }
  __builtin_amdgcn_s_wait_asynccnt(0);
  __syncthreads();
#else
  {
    const uint4* ksrc = (const uint4*)(base + (size_t)tid * (3 * D_) + koff);
    const uint4* vsrc = (const uint4*)(base + (size_t)tid * (3 * D_) + voff);
    uint4* kdst = (uint4*)&ks[tid][0];
    uint4* vdst = (uint4*)&vs[tid][0];
#pragma unroll
    for (int c = 0; c < 4; ++c) { kdst[c] = ksrc[c]; vdst[c] = vsrc[c]; }
  }
  __syncthreads();
#endif

  // Q fragments straight from global (bf16 pairs are contiguous dwords)
  FragBF qf[2];
#pragma unroll
  for (int mi = 0; mi < 2; ++mi) {
    const unsigned short* qr = base + (size_t)(w * 32 + mi * 16 + l15) * (3 * D_) + qoff;
#pragma unroll
    for (int v = 0; v < 8; ++v) {
      const int k = 2 * v + 16 * hi;
      qf[mi].u[v] = *(const unsigned int*)(qr + k);
    }
  }

  v8f cacc[2][2];
  float mrun[2][8], lrun[2][8];
#pragma unroll
  for (int mi = 0; mi < 2; ++mi)
#pragma unroll
    for (int r = 0; r < 8; ++r) {
      mrun[mi][r] = -1e30f; lrun[mi][r] = 0.f;
      cacc[mi][0][r] = 0.f; cacc[mi][1][r] = 0.f;
    }
  const float scale = 0.17677669529663687f;  // 1/sqrt(32)

  for (int kb = 0; kb < 8; ++kb) {
    // K^T fragments (B: kdim = dh, cols = keys)
    FragBF kf[2];
#pragma unroll
    for (int nj = 0; nj < 2; ++nj) {
      const int key = kb * 32 + nj * 16 + l15;
#pragma unroll
      for (int v = 0; v < 8; ++v)
        kf[nj].u[v] = *(const unsigned int*)&ks[key][2 * v + 16 * hi];
    }
    // scores
    v8f s[2][2];
#pragma unroll
    for (int mi = 0; mi < 2; ++mi)
#pragma unroll
      for (int nj = 0; nj < 2; ++nj) {
        v8f z;
#pragma unroll
        for (int r = 0; r < 8; ++r) z[r] = 0.f;
        s[mi][nj] = __builtin_amdgcn_wmma_f32_16x16x32_bf16(
            false, qf[mi].v, false, kf[nj].v, (short)0, z, false, false);
      }
    // online softmax update + stage P to LDS
#pragma unroll
    for (int mi = 0; mi < 2; ++mi) {
#pragma unroll
      for (int r = 0; r < 8; ++r) {
        float s0 = s[mi][0][r] * scale, s1 = s[mi][1][r] * scale;
        float mx = fmaxf(s0, s1);
#pragma unroll
        for (int o = 1; o < 16; o <<= 1) mx = fmaxf(mx, __shfl_xor(mx, o, 32));
        const float nm = fmaxf(mrun[mi][r], mx);
        const float al = __expf(mrun[mi][r] - nm);
        const float e0 = __expf(s0 - nm), e1 = __expf(s1 - nm);
        float rs = e0 + e1;
#pragma unroll
        for (int o = 1; o < 16; o <<= 1) rs += __shfl_xor(rs, o, 32);
        lrun[mi][r] = lrun[mi][r] * al + rs;
        mrun[mi][r] = nm;
        cacc[mi][0][r] *= al;
        cacc[mi][1][r] *= al;
        const int prow = mi * 16 + r + 8 * hi;
        Pbuf[w][prow][l15]      = f2bf(e0);
        Pbuf[w][prow][l15 + 16] = f2bf(e1);
      }
    }
    __syncthreads();  // make cross-lane P stores visible
    // P fragments (A: rows = queries, kdim = keys of this block)
    FragBF pf[2];
#pragma unroll
    for (int mi = 0; mi < 2; ++mi) {
      const int m = mi * 16 + l15;
#pragma unroll
      for (int v = 0; v < 8; ++v) {
        const int k = 2 * v + 16 * hi;
        pf[mi].u[v] = *(const unsigned int*)&Pbuf[w][m][k];
      }
    }
    // V fragments (B: kdim = keys, cols = dh) from row-major LDS
    FragBF vf[2];
#pragma unroll
    for (int nj = 0; nj < 2; ++nj) {
      const int dh = nj * 16 + l15;
#pragma unroll
      for (int v = 0; v < 8; ++v) {
        const int key = kb * 32 + 2 * v + 16 * hi;
        const unsigned int lo = vs[key][dh];
        const unsigned int hv = vs[key + 1][dh];
        vf[nj].u[v] = lo | (hv << 16);
      }
    }
#pragma unroll
    for (int mi = 0; mi < 2; ++mi)
#pragma unroll
      for (int nj = 0; nj < 2; ++nj)
        cacc[mi][nj] = __builtin_amdgcn_wmma_f32_16x16x32_bf16(
            false, pf[mi].v, false, vf[nj].v, (short)0, cacc[mi][nj], false, false);
    __syncthreads();
  }

  // normalize + store ctx (bf16)
#pragma unroll
  for (int mi = 0; mi < 2; ++mi)
#pragma unroll
    for (int nj = 0; nj < 2; ++nj)
#pragma unroll
      for (int r = 0; r < 8; ++r) {
        const int srow = w * 32 + mi * 16 + r + 8 * hi;
        const float v = cacc[mi][nj][r] / lrun[mi][r];
        ctxOut[(row0 + srow) * (size_t)D_ + h * 32 + nj * 16 + l15] = f2bf(v);
      }
}

// ---------------------------------------------------------------------------
// o = bf16(aw * sa)
// ---------------------------------------------------------------------------
__global__ __launch_bounds__(256) void mul_bf16_kernel(
    const float* __restrict__ a, const float* __restrict__ b,
    unsigned short* __restrict__ o, long long n) {
  long long i = (long long)blockIdx.x * 256 + threadIdx.x;
  if (i < n) o[i] = f2bf(a[i] * b[i]);
}

// ---------------------------------------------------------------------------
// Gating: logits = mean_s(x) @ Wg + bg ; softmax ; top-2 renormalized.
// ---------------------------------------------------------------------------
__global__ __launch_bounds__(256) void gating_kernel(
    const float* __restrict__ x, const float* __restrict__ Wg,
    const float* __restrict__ bg, float* __restrict__ gates) {
  __shared__ float xm[I_];
  __shared__ float red[256];
  const int tid = threadIdx.x;
  for (int b = 0; b < B_; ++b) {
    for (int i = tid; i < I_; i += 256) {
      float s = 0.f;
      for (int s2 = 0; s2 < S_; ++s2) s += x[((size_t)b * S_ + s2) * I_ + i];
      xm[i] = s * (1.0f / S_);
    }
    __syncthreads();
    float logit[E_];
    for (int e = 0; e < E_; ++e) {
      float p = 0.f;
      for (int i = tid; i < I_; i += 256) p += xm[i] * Wg[i * E_ + e];
      red[tid] = p; __syncthreads();
      for (int st = 128; st > 0; st >>= 1) {
        if (tid < st) red[tid] += red[tid + st];
        __syncthreads();
      }
      logit[e] = red[0] + bg[e];
      __syncthreads();
    }
    if (tid == 0) {
      float mx = logit[0];
      for (int e = 1; e < E_; ++e) mx = fmaxf(mx, logit[e]);
      float pe[E_]; float tot = 0.f;
      for (int e = 0; e < E_; ++e) { pe[e] = __expf(logit[e] - mx); tot += pe[e]; }
      for (int e = 0; e < E_; ++e) pe[e] /= tot;
      int i0 = 0;
      for (int e = 1; e < E_; ++e) if (pe[e] > pe[i0]) i0 = e;
      int i1 = -1;
      for (int e = 0; e < E_; ++e)
        if (e != i0 && (i1 < 0 || pe[e] > pe[i1])) i1 = e;
      const float sum2 = pe[i0] + pe[i1];
      for (int e = 0; e < E_; ++e) gates[b * E_ + e] = 0.f;
      gates[b * E_ + i0] = pe[i0] / sum2;
      gates[b * E_ + i1] = pe[i1] / sum2;
    }
    __syncthreads();
  }
}

// ---------------------------------------------------------------------------
// out[b,s,i] = sum_e gates[b,e] * y[e,b,s,i]
// ---------------------------------------------------------------------------
__global__ __launch_bounds__(256) void combine_kernel(
    const float* __restrict__ y, const float* __restrict__ g,
    float* __restrict__ out, int n) {
  const int i = blockIdx.x * 256 + threadIdx.x;
  if (i >= n) return;
  const int b = i >> 18;  // 256*1024 elements per batch
  float acc = 0.f;
#pragma unroll
  for (int e = 0; e < E_; ++e)
    acc += g[b * E_ + e] * y[(size_t)e * ((size_t)M_ * I_) + i];
  out[i] = acc;
}

// ---------------------------------------------------------------------------
// Host orchestration
// ---------------------------------------------------------------------------
extern "C" void kernel_launch(void* const* d_in, const int* in_sizes, int n_in,
                              void* d_out, int out_size, void* d_ws, size_t ws_size,
                              hipStream_t stream) {
  (void)in_sizes; (void)n_in; (void)out_size; (void)ws_size;
  const float* x    = (const float*)d_in[0];
  const float* Wg   = (const float*)d_in[1];
  const float* bg   = (const float*)d_in[2];
  const float* W1   = (const float*)d_in[3];
  const float* b1   = (const float*)d_in[4];
  const float* W2   = (const float*)d_in[5];
  const float* b2   = (const float*)d_in[6];
  const float* ln1g = (const float*)d_in[7];
  const float* ln1b = (const float*)d_in[8];
  const float* Wqkv = (const float*)d_in[9];
  const float* bqkv = (const float*)d_in[10];
  const float* Wo   = (const float*)d_in[11];
  const float* bo   = (const float*)d_in[12];
  const float* Wa1  = (const float*)d_in[13];
  const float* ba1  = (const float*)d_in[14];
  const float* Wa2  = (const float*)d_in[15];
  const float* ba2  = (const float*)d_in[16];
  const float* Ws1  = (const float*)d_in[17];
  const float* bs1  = (const float*)d_in[18];
  const float* Ws2  = (const float*)d_in[19];
  const float* bs2  = (const float*)d_in[20];
  const float* Wd1  = (const float*)d_in[21];
  const float* bd1  = (const float*)d_in[22];
  const float* Wd2  = (const float*)d_in[23];
  const float* bd2  = (const float*)d_in[24];
  const float* ln2g = (const float*)d_in[25];
  const float* ln2b = (const float*)d_in[26];
  float* out = (float*)d_out;

  char* ws = (char*)d_ws;
  size_t off = 0;
  auto alloc  = [&](size_t bytes) -> void* {
    void* p = ws + off;
    off += (bytes + 255) & ~(size_t)255;
    return p;
  };
  auto allocB = [&](size_t n) { return (unsigned short*)alloc(n * 2); };
  auto allocU = [&](size_t n) { return (unsigned int*)alloc(n * 4); };
  auto allocF = [&](size_t n) { return (float*)alloc(n * 4); };

  const size_t EID  = (size_t)E_ * I_ * D_;
  const size_t EDD  = (size_t)E_ * D_ * D_;
  const size_t ED3  = (size_t)E_ * D_ * 3 * D_;
  const size_t EMD  = (size_t)E_ * M_ * D_;
  const size_t EMI  = (size_t)E_ * M_ * I_;
  const size_t EM3D = (size_t)E_ * M_ * 3 * D_;

  unsigned int* W1p  = allocU(EID / 2);
  unsigned int* W2p  = allocU(EDD / 2);
  unsigned int* Wqkp = allocU(ED3 / 2);
  unsigned int* Wop  = allocU(EDD / 2);
  unsigned int* Wa1p = allocU(EDD / 2);
  unsigned int* Wa2p = allocU(EDD / 2);
  unsigned int* Ws1p = allocU(EDD / 2);
  unsigned int* Ws2p = allocU(EDD / 2);
  unsigned int* Wd1p = allocU(EDD / 2);
  unsigned int* Wd2p = allocU(EID / 2);
  unsigned short* xb   = allocB((size_t)M_ * I_);
  unsigned short* t1b  = allocB(EMD);
  float*          t2f  = allocF(EMD);
  unsigned short* hb   = allocB(EMD);
  unsigned short* qkvb = allocB(EM3D);
  unsigned short* ctxb = allocB(EMD);
  unsigned short* ab   = allocB(EMD);
  unsigned short* u1b  = allocB(EMD);
  float*          awf  = allocF(EMD);
  unsigned short* u2b  = allocB(EMD);
  float*          saf  = allocF(EMD);
  unsigned short* ob   = allocB(EMD);
  unsigned short* dtb  = allocB(EMD);
  float*          ypf  = allocF(EMI);
  float*          yf   = allocF(EMI);
  float*          gates = allocF(16);

  auto convpk = [&](const float* s, unsigned int* d, int N, size_t total) {
    const long long nPairs = (long long)(total / 2);
    f32_to_bf16pk_kernel<<<dim3((unsigned)((nPairs + 255) / 256)), dim3(256), 0, stream>>>(
        s, d, N, nPairs);
  };
  convpk(W1,  W1p,  D_,     EID);
  convpk(W2,  W2p,  D_,     EDD);
  convpk(Wqkv, Wqkp, 3 * D_, ED3);
  convpk(Wo,  Wop,  D_,     EDD);
  convpk(Wa1, Wa1p, D_,     EDD);
  convpk(Wa2, Wa2p, D_,     EDD);
  convpk(Ws1, Ws1p, D_,     EDD);
  convpk(Ws2, Ws2p, D_,     EDD);
  convpk(Wd1, Wd1p, D_,     EDD);
  convpk(Wd2, Wd2p, I_,     EID);
  f32_to_bf16_kernel<<<dim3((unsigned)(((size_t)M_ * I_ + 255) / 256)), dim3(256), 0, stream>>>(
      x, xb, (long long)M_ * I_);

  const dim3 blk(256);
  const long long sMD = (long long)M_ * D_;

  // t1 = relu(x @ W1 + b1) -> bf16
  gemm_bf16_wmma<1><<<dim3(D_ / 128, M_ / 128, E_), blk, 0, stream>>>(
      xb, 0, W1p, b1, nullptr, t1b, M_, D_, I_);
  // t2 = t1 @ W2 + b2 -> f32
  gemm_bf16_wmma<0><<<dim3(D_ / 128, M_ / 128, E_), blk, 0, stream>>>(
      t1b, sMD, W2p, b2, t2f, nullptr, M_, D_, D_);
  // h = LN1(t2) -> bf16
  layernorm_kernel<true><<<dim3(E_ * M_), blk, 0, stream>>>(
      t2f, ln1g, ln1b, nullptr, hb, D_, M_);
  // qkv = h @ Wqkv + bqkv -> bf16
  gemm_bf16_wmma<2><<<dim3(3 * D_ / 128, M_ / 128, E_), blk, 0, stream>>>(
      hb, sMD, Wqkp, bqkv, nullptr, qkvb, M_, 3 * D_, D_);
  // attention -> ctx bf16
  attention_kernel<<<dim3(H_, B_, E_), blk, 0, stream>>>(qkvb, ctxb);
  // a = ctx @ Wo + bo -> bf16
  gemm_bf16_wmma<2><<<dim3(D_ / 128, M_ / 128, E_), blk, 0, stream>>>(
      ctxb, sMD, Wop, bo, nullptr, ab, M_, D_, D_);
  // u1 = relu(a @ Wa1 + ba1)
  gemm_bf16_wmma<1><<<dim3(D_ / 128, M_ / 128, E_), blk, 0, stream>>>(
      ab, sMD, Wa1p, ba1, nullptr, u1b, M_, D_, D_);
  // aw = sigmoid(u1 @ Wa2 + ba2) -> f32
  gemm_bf16_wmma<3><<<dim3(D_ / 128, M_ / 128, E_), blk, 0, stream>>>(
      u1b, sMD, Wa2p, ba2, awf, nullptr, M_, D_, D_);
  // u2 = relu(a @ Ws1 + bs1)
  gemm_bf16_wmma<1><<<dim3(D_ / 128, M_ / 128, E_), blk, 0, stream>>>(
      ab, sMD, Ws1p, bs1, nullptr, u2b, M_, D_, D_);
  // sa = tanh(u2 @ Ws2 + bs2) -> f32
  gemm_bf16_wmma<4><<<dim3(D_ / 128, M_ / 128, E_), blk, 0, stream>>>(
      u2b, sMD, Ws2p, bs2, saf, nullptr, M_, D_, D_);
  // o = aw * sa -> bf16
  mul_bf16_kernel<<<dim3((unsigned)((EMD + 255) / 256)), blk, 0, stream>>>(
      awf, saf, ob, (long long)EMD);
  // dt = relu(o @ Wd1 + bd1)
  gemm_bf16_wmma<1><<<dim3(D_ / 128, M_ / 128, E_), blk, 0, stream>>>(
      ob, sMD, Wd1p, bd1, nullptr, dtb, M_, D_, D_);
  // ypre = dt @ Wd2 + bd2 -> f32
  gemm_bf16_wmma<0><<<dim3(I_ / 128, M_ / 128, E_), blk, 0, stream>>>(
      dtb, sMD, Wd2p, bd2, ypf, nullptr, M_, I_, D_);
  // y = LN2(ypre) -> f32
  layernorm_kernel<false><<<dim3(E_ * M_), blk, 0, stream>>>(
      ypf, ln2g, ln2b, yf, nullptr, I_, M_);
  // gates
  gating_kernel<<<dim3(1), blk, 0, stream>>>(x, Wg, bg, gates);
  // out = sum_e gates * y
  combine_kernel<<<dim3((unsigned)((M_ * I_ + 255) / 256)), blk, 0, stream>>>(
      yf, gates, out, M_ * I_);
}
